// DiGressNet_GTv1_48490180772346
// MI455X (gfx1250) — compile-verified, matrix operand order checked by reading, not run on hardware
//
#include <hip/hip_runtime.h>
#include <stdint.h>
#include <stddef.h>

// ---------------------------------------------------------------------------
// Types for CDNA5 WMMA (wave32, 16x16x32 bf16 -> f32)
// ---------------------------------------------------------------------------
typedef unsigned short u16;
typedef __bf16 bf16;
typedef bf16  v16bf __attribute__((ext_vector_type(16)));
typedef float v8f   __attribute__((ext_vector_type(8)));
typedef unsigned int u32x4 __attribute__((ext_vector_type(4)));

union Frag { v16bf bf; u32x4 q[2]; };

__device__ __forceinline__ u16 f2bf(float f) {
    union { float f; unsigned u; } c{f};
    unsigned r = c.u + 0x7FFFu + ((c.u >> 16) & 1u);   // round-to-nearest-even
    return (u16)(r >> 16);
}
__device__ __forceinline__ float bf2f(u16 h) {
    union { unsigned u; float f; } c{ ((unsigned)h) << 16 };
    return c.f;
}

// Load one 16x32 bf16 WMMA fragment (A row-major / B from pre-transposed W).
// p = row base (row*ld already applied) + k0 ; sel = (lane>>4)*8 elements.
__device__ __forceinline__ v16bf load_frag(const u16* p, int sel) {
    Frag f;
    f.q[0] = *(const u32x4*)(p + sel);
    f.q[1] = *(const u32x4*)(p + 16 + sel);
    return f.bf;
}

#define WMMA_BF16(A, B, C) \
    __builtin_amdgcn_wmma_f32_16x16x32_bf16(false, (A), false, (B), (short)0, (C), false, false)

// ---------------------------------------------------------------------------
// Parameter layout (floats) — jax tree_flatten order: sorted dict keys,
// lists in order, {'b','w'} -> b first; {'b','g'} -> b first.
// ---------------------------------------------------------------------------
constexpr size_t P_INE0_B = 0,       P_INE0_W = 128;
constexpr size_t P_INE1_B = 768,     P_INE1_W = 1024;
constexpr size_t P_INX0_B = 33792,   P_INX0_W = 34048;
constexpr size_t P_INX1_B = 41216,   P_INX1_W = 41472;
constexpr size_t P_L0 = 107008, P_LS = 1579520;
// per-layer relative offsets (sorted: E,K,LNe,LNe2,LNx,LNx2,MLPe,MLPx,Ni,Nj,Q,V,WOe,WOx)
constexpr size_t PL_E_B=0,        PL_E_W=256;
constexpr size_t PL_K_B=65792,    PL_K_W=66048;
constexpr size_t PL_LNE_B=131584, PL_LNE_G=131840;
constexpr size_t PL_LNE2_B=132096,PL_LNE2_G=132352;
constexpr size_t PL_LNX_B=132608, PL_LNX_G=132864;
constexpr size_t PL_LNX2_B=133120,PL_LNX2_G=133376;
constexpr size_t PL_ME0_B=133632, PL_ME0_W=134656;
constexpr size_t PL_ME1_B=396800, PL_ME1_W=397056;
constexpr size_t PL_MX0_B=659200, PL_MX0_W=660224;
constexpr size_t PL_MX1_B=922368, PL_MX1_W=922624;
constexpr size_t PL_NI_B=1184768, PL_NI_W=1185024;
constexpr size_t PL_NJ_B=1250560, PL_NJ_W=1250816;
constexpr size_t PL_Q_B=1316352,  PL_Q_W=1316608;
constexpr size_t PL_V_B=1382144,  PL_V_W=1382400;
constexpr size_t PL_WOE_B=1447936,PL_WOE_W=1448192;
constexpr size_t PL_WOX_B=1513728,PL_WOX_W=1513984;
constexpr size_t P_OUTE0_B=6425088, P_OUTE0_W=6425216;
constexpr size_t P_OUTE1_B=6457984, P_OUTE1_W=6457989;
constexpr size_t P_OUTX0_B=6458629, P_OUTX0_W=6458885;
constexpr size_t P_OUTX1_B=6524421, P_OUTX1_W=6524437;

// ---------------------------------------------------------------------------
// Weight table: f32 (din,dout) row-major  ->  bf16 transposed (Np x Kp),
// zero padded; biases padded to Np as f32.
// ---------------------------------------------------------------------------
struct WDesc { unsigned srcW, srcB, din, dout, Kp, Np, dstW, dstB; };
struct WTab  { WDesc d[56]; unsigned wbytes, bbytes; };
constexpr unsigned rupc(unsigned v, unsigned m) { return (v + m - 1) / m * m; }

constexpr WTab makeWTab() {
    WTab t{}; unsigned off = 0, boff = 0; int n = 0;
    auto add = [&](size_t sw, size_t sb, unsigned di, unsigned dou) {
        WDesc w{}; w.srcW=(unsigned)sw; w.srcB=(unsigned)sb; w.din=di; w.dout=dou;
        w.Kp = rupc(di, 32); w.Np = rupc(dou, 64);
        w.dstW = off;  off  += w.Np * w.Kp * 2u;
        w.dstB = boff; boff += w.Np * 4u;
        t.d[n++] = w;
    };
    add(P_INE0_W, P_INE0_B, 5, 128);
    add(P_INE1_W, P_INE1_B, 128, 256);
    add(P_INX0_W, P_INX0_B, 28, 256);
    add(P_INX1_W, P_INX1_B, 256, 256);
    for (int l = 0; l < 4; l++) {
        size_t B0 = P_L0 + (size_t)l * P_LS;
        add(B0+PL_Q_W,  B0+PL_Q_B,  256, 256);
        add(B0+PL_K_W,  B0+PL_K_B,  256, 256);
        add(B0+PL_V_W,  B0+PL_V_B,  256, 256);
        add(B0+PL_NI_W, B0+PL_NI_B, 256, 256);
        add(B0+PL_NJ_W, B0+PL_NJ_B, 256, 256);
        add(B0+PL_E_W,  B0+PL_E_B,  256, 256);
        add(B0+PL_WOX_W,B0+PL_WOX_B,256, 256);
        add(B0+PL_WOE_W,B0+PL_WOE_B,256, 256);
        add(B0+PL_ME0_W,B0+PL_ME0_B,256, 1024);
        add(B0+PL_ME1_W,B0+PL_ME1_B,1024,256);
        add(B0+PL_MX0_W,B0+PL_MX0_B,256, 1024);
        add(B0+PL_MX1_W,B0+PL_MX1_B,1024,256);
    }
    add(P_OUTX0_W, P_OUTX0_B, 256, 256);
    add(P_OUTX1_W, P_OUTX1_B, 256, 16);
    add(P_OUTE0_W, P_OUTE0_B, 256, 128);
    add(P_OUTE1_W, P_OUTE1_B, 128, 5);
    t.wbytes = off; t.bbytes = boff;
    return t;
}
constexpr WTab TAB = makeWTab();             // host use
static __constant__ WTab d_TAB = makeWTab(); // device use

// per-layer weight slot indices inside TAB (base 4 + l*12 + which)
enum { LQ=0, LK, LV, LNI, LNJ, LE_, LWOX, LWOE, LME0, LME1, LMX0, LMX1 };
constexpr int W_INE0=0, W_INE1=1, W_INX0=2, W_INX1=3;
constexpr int W_OUTX0=52, W_OUTX1=53, W_OUTE0=54, W_OUTE1=55;

// ---------------------------------------------------------------------------
// Workspace layout (bytes)
// ---------------------------------------------------------------------------
constexpr size_t AL256(size_t x) { return (x + 255) & ~(size_t)255; }
constexpr size_t WS_WT   = 0;
constexpr size_t WS_BIAS = AL256(TAB.wbytes);
constexpr size_t WS_EF   = AL256(WS_BIAS + TAB.bbytes);       // E residual f32   67.1 MB
constexpr size_t WS_EBF  = WS_EF   + 67108864;                // LN(E) bf16       33.5 MB
constexpr size_t WS_ETMP = WS_EBF  + 33554432;                // embed tmp f32 (reused: e_new bf16 + att)
constexpr size_t WS_ENEW = WS_ETMP;                           // e_new bf16       33.5 MB
constexpr size_t WS_ATT  = WS_ETMP + 33554432;                // att f32           2.1 MB
constexpr size_t WS_EIN  = WS_ETMP + 67108864;                // padded edge in bf16 4 MB
constexpr size_t WS_EH1  = WS_EIN  + 4194304;                 // edge hidden bf16 (reused as EoH) 16.8 MB
constexpr size_t WS_XF   = WS_EH1  + 16777216;                // x residual f32   512 KB
constexpr size_t WS_XBF  = WS_XF   + 524288;
constexpr size_t WS_XBF2 = WS_XBF  + 262144;
constexpr size_t WS_XOBF = WS_XBF2 + 262144;
constexpr size_t WS_XIN  = WS_XOBF + 262144;
constexpr size_t WS_XH1  = WS_XIN  + 32768;                   // (reused as XoH)
constexpr size_t WS_QF   = WS_XH1  + 262144;
constexpr size_t WS_KF   = WS_QF   + 524288;
constexpr size_t WS_VF   = WS_KF   + 524288;
constexpr size_t WS_NIF  = WS_VF   + 524288;
constexpr size_t WS_NJF  = WS_NIF  + 524288;
constexpr size_t WS_TF   = WS_NJF  + 524288;                  // out_E pre-sym f32 1.3 MB

// ---------------------------------------------------------------------------
// Kernels
// ---------------------------------------------------------------------------

// Convert/transpose/pad one weight per blockIdx.y into bf16, plus padded bias.
__global__ void k_convert(const float* __restrict__ P, u16* __restrict__ wt,
                          float* __restrict__ bias) {
    WDesc w = d_TAB.d[blockIdx.y];
    if (blockIdx.x == 0)
        for (unsigned n = threadIdx.x; n < w.Np; n += 256)
            bias[w.dstB / 4 + n] = (n < w.dout) ? P[w.srcB + n] : 0.f;
    size_t total = (size_t)w.Np * w.Kp;
    for (size_t idx = (size_t)blockIdx.x * 256 + threadIdx.x; idx < total;
         idx += (size_t)gridDim.x * 256) {
        unsigned n = (unsigned)(idx / w.Kp), k = (unsigned)(idx % w.Kp);
        float v = (n < w.dout && k < w.din) ? P[w.srcW + (size_t)k * w.dout + n] : 0.f;
        wt[w.dstW / 2 + idx] = f2bf(v);
    }
}

// Generic fused GEMM: OUT[R x N] = act( A[R x K](bf16) @ W^T + bias [+NiNj] [+res] )
// One wave computes a 32x64 C block (8 accumulators) so each k-step issues
// 8 v_wmma against 12 b128 loads (B fragments amortized over 2 row tiles).
__global__ __launch_bounds__(32) void k_gemm(
    const u16* __restrict__ A, const u16* __restrict__ W,
    const float* __restrict__ bias, int K, int N,
    const float* __restrict__ res, int ldr,
    float* __restrict__ outF, int ldf,
    u16* __restrict__ outB, int ldb2,
    int relu, const float* __restrict__ eNi, const float* __restrict__ eNj) {
    int lane = threadIdx.x;
    int row0 = blockIdx.x * 32, n0 = blockIdx.y * 64;
    int lm = lane & 15, sel = (lane >> 4) * 8, mb = (lane >> 4) * 8;
    v8f acc[2][4] = {};
    const u16* aRow0 = A + (size_t)(row0 + lm) * K;
    const u16* aRow1 = A + (size_t)(row0 + 16 + lm) * K;
    for (int k0 = 0; k0 < K; k0 += 32) {
        v16bf fa0 = load_frag(aRow0 + k0, sel);
        v16bf fa1 = load_frag(aRow1 + k0, sel);
#pragma unroll
        for (int t = 0; t < 4; t++) {
            const u16* wRow = W + (size_t)(n0 + t * 16 + lm) * K;
            v16bf fb = load_frag(wRow + k0, sel);
            acc[0][t] = WMMA_BF16(fa0, fb, acc[0][t]);
            acc[1][t] = WMMA_BF16(fa1, fb, acc[1][t]);
        }
    }
#pragma unroll
    for (int r = 0; r < 2; r++) {
#pragma unroll
        for (int t = 0; t < 4; t++) {
            int n = n0 + t * 16 + lm;
            if (n >= N) continue;
            float bv = bias ? bias[n] : 0.f;
#pragma unroll
            for (int v = 0; v < 8; v++) {
                int row = row0 + r * 16 + mb + v;
                float val = acc[r][t][v] + bv;
                if (eNi) {  // edge-bias: row = ((b*128+i)*128+j)
                    int bb = row >> 14, i = (row >> 7) & 127, j = row & 127;
                    val += eNi[(size_t)(bb * 128 + i) * 256 + n] +
                           eNj[(size_t)(bb * 128 + j) * 256 + n];
                }
                if (res) val += res[(size_t)row * ldr + n];
                if (relu) val = val > 0.f ? val : 0.f;
                if (outF) outF[(size_t)row * ldf + n] = val;
                if (outB) outB[(size_t)row * ldb2 + n] = f2bf(val);
            }
        }
    }
}

// Fused MLP: out = A(bf16,Rx256) @ W0^T(1024x256) +b0 -> leaky -> @ W1^T(256x1024)
//            + b1 + res -> f32 (+bf16).  Hidden tile lives in LDS only.
__global__ __launch_bounds__(32) void k_mlp(
    const u16* __restrict__ A, const u16* __restrict__ W0, const float* __restrict__ b0,
    const u16* __restrict__ W1, const float* __restrict__ b1,
    const float* __restrict__ res, float* __restrict__ outF, u16* __restrict__ outB) {
    __shared__ u16 hid[16 * 1024];   // 32 KB of the 320 KB WGP LDS
    int lane = threadIdx.x;
    int row0 = blockIdx.x * 16;
    int lm = lane & 15, sel = (lane >> 4) * 8, mb = (lane >> 4) * 8;
    Frag fa[8];
    const u16* aRow = A + (size_t)(row0 + lm) * 256;
#pragma unroll
    for (int ks = 0; ks < 8; ks++) {
        fa[ks].q[0] = *(const u32x4*)(aRow + ks * 32 + sel);
        fa[ks].q[1] = *(const u32x4*)(aRow + ks * 32 + 16 + sel);
    }
    for (int n0 = 0; n0 < 1024; n0 += 16) {
        v8f acc{};
        const u16* wRow = W0 + (size_t)(n0 + lm) * 256;
#pragma unroll
        for (int ks = 0; ks < 8; ks++) {
            v16bf fb = load_frag(wRow + ks * 32, sel);
            acc = WMMA_BF16(fa[ks].bf, fb, acc);
        }
        int n = n0 + lm; float bv = b0[n];
#pragma unroll
        for (int v = 0; v < 8; v++) {
            float val = acc[v] + bv;
            val = val > 0.f ? val : 0.01f * val;   // leaky_relu(0.01)
            hid[(mb + v) * 1024 + n] = f2bf(val);
        }
    }
    __syncthreads();
    const u16* hRow = hid + (size_t)lm * 1024;
    for (int g = 0; g < 4; g++) {
        v8f acc[4] = {};
        for (int ks = 0; ks < 32; ks++) {
            v16bf fA = load_frag(hRow + ks * 32, sel);
#pragma unroll
            for (int t = 0; t < 4; t++) {
                const u16* wRow = W1 + (size_t)(g * 64 + t * 16 + lm) * 1024;
                v16bf fb = load_frag(wRow + ks * 32, sel);
                acc[t] = WMMA_BF16(fA, fb, acc[t]);
            }
        }
#pragma unroll
        for (int t = 0; t < 4; t++) {
            int n = g * 64 + t * 16 + lm; float bv = b1[n];
#pragma unroll
            for (int v = 0; v < 8; v++) {
                int row = row0 + mb + v;
                float val = acc[t][v] + bv + res[(size_t)row * 256 + n];
                outF[(size_t)row * 256 + n] = val;
                if (outB) outB[(size_t)row * 256 + n] = f2bf(val);
            }
        }
    }
}

// LayerNorm over D=256; one wave per row; population variance, eps=1e-5.
__global__ __launch_bounds__(256) void k_ln(const float* __restrict__ in,
                                            const float* __restrict__ g,
                                            const float* __restrict__ b,
                                            float* outF, u16* outB, int R) {
    int wid = threadIdx.x >> 5, lane = threadIdx.x & 31;
    int row = blockIdx.x * 8 + wid;
    if (row >= R) return;
    const float* x = in + (size_t)row * 256;
    float v[8], s = 0.f, sq = 0.f;
#pragma unroll
    for (int t = 0; t < 8; t++) { v[t] = x[t * 32 + lane]; s += v[t]; sq += v[t] * v[t]; }
#pragma unroll
    for (int m = 16; m >= 1; m >>= 1) { s += __shfl_xor(s, m, 32); sq += __shfl_xor(sq, m, 32); }
    float mean = s * (1.f / 256.f);
    float var  = sq * (1.f / 256.f) - mean * mean;
    float rstd = rsqrtf(var + 1e-5f);
#pragma unroll
    for (int t = 0; t < 8; t++) {
        int c = t * 32 + lane;
        float o = (v[t] - mean) * rstd * g[c] + b[c];
        if (outF) outF[(size_t)row * 256 + c] = o;
        if (outB) outB[(size_t)row * 256 + c] = f2bf(o);
    }
}

// att[b,i,j,h] = sum_d Q[b,i,h,d]*e_new[b,i,j,h,d]*K[b,j,h,d] / sqrt(32)
__global__ __launch_bounds__(256) void k_attn_scores(const u16* __restrict__ E,
                                                     const float* __restrict__ Q,
                                                     const float* __restrict__ Kf,
                                                     float* __restrict__ att) {
    int idx = blockIdx.x * 256 + threadIdx.x;            // 524288 total
    int h = idx & 7, j = (idx >> 3) & 127, i = (idx >> 10) & 127, b = idx >> 17;
    const u16*  e = E  + ((((size_t)b * 128 + i) * 128 + j) << 8) + h * 32;
    const float* q = Q  + (((size_t)b * 128 + i) << 8) + h * 32;
    const float* k = Kf + (((size_t)b * 128 + j) << 8) + h * 32;
    float acc = 0.f;
#pragma unroll
    for (int t = 0; t < 4; t++) {
        u32x4 ev = *(const u32x4*)(e + t * 8);
#pragma unroll
        for (int u = 0; u < 4; u++) {
            int d = t * 8 + u * 2; unsigned w = ev[u];
            acc += q[d] * k[d] * bf2f((u16)w) + q[d + 1] * k[d + 1] * bf2f((u16)(w >> 16));
        }
    }
    att[idx] = acc * 0.17677669529663687f;               // 1/sqrt(32)
}

// softmax over i (axis=1) for each (b,j,h) column; one wave per column.
__global__ __launch_bounds__(256) void k_attn_softmax(float* __restrict__ att) {
    int wid = threadIdx.x >> 5, lane = threadIdx.x & 31;
    int c = blockIdx.x * 8 + wid;                        // 4096 columns
    int b = c >> 10, rem = c & 1023;                     // rem = j*8+h
    float* p = att + (size_t)b * 131072 + rem;
    float v[4], m = -1e30f;
#pragma unroll
    for (int t = 0; t < 4; t++) { v[t] = p[(size_t)(lane + t * 32) * 1024]; m = fmaxf(m, v[t]); }
#pragma unroll
    for (int s = 16; s >= 1; s >>= 1) m = fmaxf(m, __shfl_xor(m, s, 32));
    float sum = 0.f;
#pragma unroll
    for (int t = 0; t < 4; t++) { v[t] = __expf(v[t] - m); sum += v[t]; }
#pragma unroll
    for (int s = 16; s >= 1; s >>= 1) sum += __shfl_xor(sum, s, 32);
    float inv = 1.f / sum;
#pragma unroll
    for (int t = 0; t < 4; t++) p[(size_t)(lane + t * 32) * 1024] = v[t] * inv;
}

// xo[b,i,h,d] = sum_j att[b,i,j,h] * V[b,j,h,d]   -> bf16
__global__ __launch_bounds__(256) void k_attn_apply(const float* __restrict__ att,
                                                    const float* __restrict__ V,
                                                    u16* __restrict__ xo) {
    int idx = blockIdx.x * 256 + threadIdx.x;            // 131072 total
    int d = idx & 31, h = (idx >> 5) & 7, i = (idx >> 8) & 127, b = idx >> 15;
    const float* a = att + (size_t)b * 131072 + (size_t)i * 1024 + h;
    const float* v = V + (((size_t)b * 128) << 8) + h * 32 + d;
    float acc = 0.f;
    for (int j = 0; j < 128; j++) acc += a[(size_t)j * 8] * v[(size_t)j * 256];
    xo[(((size_t)b * 128 + i) << 8) + h * 32 + d] = f2bf(acc);
}

// pack X=concat(noise_x,extra_x) into 512x32 bf16 (zero padded 28->32)
__global__ void k_pack_x(const float* __restrict__ nx, const float* __restrict__ ex,
                         u16* __restrict__ out) {
    int idx = blockIdx.x * 256 + threadIdx.x;
    if (idx >= 512 * 32) return;
    int r = idx >> 5, c = idx & 31;
    float v = c < 16 ? nx[r * 16 + c] : (c < 28 ? ex[r * 12 + (c - 16)] : 0.f);
    out[idx] = f2bf(v);
}
// pack noise_e (..,5) into 65536x32 bf16
__global__ void k_pack_e(const float* __restrict__ ne, u16* __restrict__ out) {
    size_t idx = (size_t)blockIdx.x * 256 + threadIdx.x;
    size_t r = idx >> 5; int c = (int)(idx & 31);
    out[idx] = f2bf(c < 5 ? ne[r * 5 + c] : 0.f);
}
// E = (T + T^T)/2  (full 256 channels, f32)
__global__ void k_sym_full(const float* __restrict__ t, float* __restrict__ e) {
    size_t idx = (size_t)blockIdx.x * 256 + threadIdx.x;
    int c = (int)(idx & 255); size_t r = idx >> 8;
    int j = (int)(r & 127), i = (int)((r >> 7) & 127), b = (int)(r >> 14);
    size_t tix = (((size_t)(b * 128 + j) * 128) + i) * 256 + c;
    e[idx] = 0.5f * (t[idx] + t[tix]);
}
// Eo = (T + T^T)/2  (5 channels)
__global__ void k_sym_out(const float* __restrict__ t, float* __restrict__ o) {
    int idx = blockIdx.x * 256 + threadIdx.x;
    if (idx >= 327680) return;
    int c = idx % 5, r = idx / 5;
    int j = r & 127, i = (r >> 7) & 127, b = r >> 14;
    int tix = ((b * 128 + j) * 128 + i) * 5 + c;
    o[idx] = 0.5f * (t[idx] + t[tix]);
}

// ---------------------------------------------------------------------------
// Host orchestration
// ---------------------------------------------------------------------------
extern "C" void kernel_launch(void* const* d_in, const int* in_sizes, int n_in,
                              void* d_out, int out_size, void* d_ws, size_t ws_size,
                              hipStream_t stream) {
    (void)in_sizes; (void)n_in; (void)out_size; (void)ws_size;
    const float* nx = (const float*)d_in[0];   // (4,128,16)
    const float* ne = (const float*)d_in[1];   // (4,128,128,5)
    const float* ex = (const float*)d_in[2];   // (4,128,12)
    const float* Pf = (const float*)d_in[5];   // flattened params
    char* ws = (char*)d_ws;
    auto F  = [&](size_t o) { return (float*)(ws + o); };
    auto U  = [&](size_t o) { return (u16*)(ws + o); };
    auto wt = [&](int i) { return (const u16*)(ws + WS_WT + TAB.d[i].dstW); };
    auto bi = [&](int i) { return (const float*)(ws + WS_BIAS + TAB.d[i].dstB); };
    auto WL = [&](int l, int w) { return 4 + l * 12 + w; };
    auto gemm = [&](const u16* A, int wi, int R, int K, int N,
                    const float* res, int ldr, float* oF, int ldf,
                    u16* oB, int ldb2, int relu,
                    const float* ni, const float* nj) {
        dim3 grid(R / 32, (N + 63) / 64);
        k_gemm<<<grid, 32, 0, stream>>>(A, wt(wi), bi(wi), K, N, res, ldr,
                                        oF, ldf, oB, ldb2, relu, ni, nj);
    };

    // --- weight conversion (f32 -> transposed padded bf16) ---
    k_convert<<<dim3(1024, 56), 256, 0, stream>>>(Pf, U(WS_WT), F(WS_BIAS));

    // --- embeddings ---
    k_pack_x<<<64, 256, 0, stream>>>(nx, ex, U(WS_XIN));
    k_pack_e<<<8192, 256, 0, stream>>>(ne, U(WS_EIN));
    gemm(U(WS_XIN), W_INX0, 512, 32, 256, nullptr, 0, nullptr, 0, U(WS_XH1), 256, 1, nullptr, nullptr);
    gemm(U(WS_XH1), W_INX1, 512, 256, 256, nullptr, 0, F(WS_XF), 256, nullptr, 0, 1, nullptr, nullptr);
    gemm(U(WS_EIN), W_INE0, 65536, 32, 128, nullptr, 0, nullptr, 0, U(WS_EH1), 128, 1, nullptr, nullptr);
    gemm(U(WS_EH1), W_INE1, 65536, 128, 256, nullptr, 0, F(WS_ETMP), 256, nullptr, 0, 1, nullptr, nullptr);
    k_sym_full<<<65536, 256, 0, stream>>>(F(WS_ETMP), F(WS_EF));

    // --- transformer layers ---
    for (int l = 0; l < 4; l++) {
        size_t LB = P_L0 + (size_t)l * P_LS;
        // LNx / LNe (LN output becomes the residual stream: f32 in-place + bf16)
        k_ln<<<64, 256, 0, stream>>>(F(WS_XF), Pf + LB + PL_LNX_G, Pf + LB + PL_LNX_B,
                                     F(WS_XF), U(WS_XBF), 512);
        k_ln<<<8192, 256, 0, stream>>>(F(WS_EF), Pf + LB + PL_LNE_G, Pf + LB + PL_LNE_B,
                                       F(WS_EF), U(WS_EBF), 65536);
        // node projections
        gemm(U(WS_XBF), WL(l, LQ),  512, 256, 256, nullptr, 0, F(WS_QF), 256, nullptr, 0, 0, nullptr, nullptr);
        gemm(U(WS_XBF), WL(l, LK),  512, 256, 256, nullptr, 0, F(WS_KF), 256, nullptr, 0, 0, nullptr, nullptr);
        gemm(U(WS_XBF), WL(l, LV),  512, 256, 256, nullptr, 0, F(WS_VF), 256, nullptr, 0, 0, nullptr, nullptr);
        gemm(U(WS_XBF), WL(l, LNI), 512, 256, 256, nullptr, 0, F(WS_NIF), 256, nullptr, 0, 0, nullptr, nullptr);
        gemm(U(WS_XBF), WL(l, LNJ), 512, 256, 256, nullptr, 0, F(WS_NJF), 256, nullptr, 0, 0, nullptr, nullptr);
        // e_new = E-proj + bias + Ni[i] + Nj[j]  (bf16)
        gemm(U(WS_EBF), WL(l, LE_), 65536, 256, 256, nullptr, 0, nullptr, 0,
             U(WS_ENEW), 256, 0, F(WS_NIF), F(WS_NJF));
        // attention
        k_attn_scores<<<2048, 256, 0, stream>>>(U(WS_ENEW), F(WS_QF), F(WS_KF), F(WS_ATT));
        k_attn_softmax<<<512, 256, 0, stream>>>(F(WS_ATT));
        k_attn_apply<<<512, 256, 0, stream>>>(F(WS_ATT), F(WS_VF), U(WS_XOBF));
        // output projections + residual
        gemm(U(WS_XOBF), WL(l, LWOX), 512, 256, 256, F(WS_XF), 256, F(WS_XF), 256,
             nullptr, 0, 0, nullptr, nullptr);
        gemm(U(WS_ENEW), WL(l, LWOE), 65536, 256, 256, F(WS_EF), 256, F(WS_EF), 256,
             nullptr, 0, 0, nullptr, nullptr);
        // MLPs (LDS-fused, leaky 0.01) with residual; bf16 mirror kept for heads
        k_ln<<<64, 256, 0, stream>>>(F(WS_XF), Pf + LB + PL_LNX2_G, Pf + LB + PL_LNX2_B,
                                     nullptr, U(WS_XBF2), 512);
        k_mlp<<<32, 32, 0, stream>>>(U(WS_XBF2), wt(WL(l, LMX0)), bi(WL(l, LMX0)),
                                     wt(WL(l, LMX1)), bi(WL(l, LMX1)),
                                     F(WS_XF), F(WS_XF), U(WS_XBF));
        k_ln<<<8192, 256, 0, stream>>>(F(WS_EF), Pf + LB + PL_LNE2_G, Pf + LB + PL_LNE2_B,
                                       nullptr, U(WS_EBF), 65536);
        k_mlp<<<4096, 32, 0, stream>>>(U(WS_EBF), wt(WL(l, LME0)), bi(WL(l, LME0)),
                                       wt(WL(l, LME1)), bi(WL(l, LME1)),
                                       F(WS_EF), F(WS_EF), U(WS_EBF));
    }

    // --- output heads ---
    float* Xo = (float*)d_out;            // 512*16
    float* Eo = (float*)d_out + 8192;     // 65536*5
    gemm(U(WS_XBF), W_OUTX0, 512, 256, 256, nullptr, 0, nullptr, 0, U(WS_XH1), 256, 1, nullptr, nullptr);
    gemm(U(WS_XH1), W_OUTX1, 512, 256, 16, nx, 16, Xo, 16, nullptr, 0, 0, nullptr, nullptr);
    gemm(U(WS_EBF), W_OUTE0, 65536, 256, 128, nullptr, 0, nullptr, 0, U(WS_EH1), 128, 1, nullptr, nullptr);
    gemm(U(WS_EH1), W_OUTE1, 65536, 128, 5, ne, 5, F(WS_TF), 5, nullptr, 0, 0, nullptr, nullptr);
    k_sym_out<<<1280, 256, 0, stream>>>(F(WS_TF), Eo);
}
